// GRU_31284541784545
// MI455X (gfx1250) — matrix operand order repaired and checked by reading, not compiled, not run on hardware
//
#include <hip/hip_runtime.h>
#include <hip/hip_bf16.h>

// ---------------------------------------------------------------------------
// GRU (2 layers, H=64) + MLP head for MI455X (gfx1250, wave32, WMMA).
// Persistent per-16-batch-rows recurrence; input projection fused with hidden
// projection (12 v_wmma per wave per step); weights VGPR-resident across the
// whole T loop; h double-buffered in LDS (f16) + fp32 master copy in VGPRs;
// one barrier per timestep; branch-free fast sigmoid/tanh (v_exp + v_rcp).
// ---------------------------------------------------------------------------

typedef __attribute__((ext_vector_type(16))) _Float16 v16h;
typedef __attribute__((ext_vector_type(8)))  _Float16 v8h;
typedef __attribute__((ext_vector_type(8)))  float    v8f;

#define GRU_B 256
#define GRU_T 1800
#define GRU_F 50
#define GRU_H 64   // padded K for layer-0 input is also 64

__device__ __forceinline__ v8f wmma_f16(v16h a, v16h b, v8f c) {
  // v_wmma_f32_16x16x32_f16  (args: neg_a, A, neg_b, B, c_mod, C, reuseA, reuseB)
  return __builtin_amdgcn_wmma_f32_16x16x32_f16(false, a, false, b, (short)0, c,
                                                false, false);
}

// Branch-free activations: one v_exp_f32 + one v_rcp_f32 each.
__device__ __forceinline__ float fast_sigmoid(float x) {
  return __builtin_amdgcn_rcpf(1.0f + __expf(-x));
}
__device__ __forceinline__ float fast_tanh(float x) {
  // tanh(x) = 1 - 2/(e^{2x}+1); saturates to +/-1 for large |x|.
  return 1.0f - 2.0f * __builtin_amdgcn_rcpf(1.0f + __expf(2.0f * x));
}

// A-matrix fragment (16x32, f16), source rows have stride 64 halves.
// Lane l (l<16): row l, K {k0..k0+8) and {k0+16..k0+24); lane l+16: +8.
template <typename PT>
__device__ __forceinline__ v16h frag_a(PT base, int k0, int lane) {
  const int r  = lane & 15;
  const int c0 = k0 + ((lane >> 4) << 3);
  union { v16h v; v8h h[2]; } u;
  u.h[0] = *(const v8h*)(base + r * 64 + c0);
  u.h[1] = *(const v8h*)(base + r * 64 + c0 + 16);
  return u.v;
}

// B-matrix fragment (32x16, f16) from row-major weight [192][64]:
// lane n (<16): column n0+n, K {k0..k0+8) and {k0+16..k0+24); lane n+16: +8.
__device__ __forceinline__ v16h frag_b(const _Float16* __restrict__ w,
                                       int n0, int k0, int lane) {
  const int n  = n0 + (lane & 15);
  const int kk = k0 + ((lane >> 4) << 3);
  union { v16h v; v8h h[2]; } u;
  u.h[0] = *(const v8h*)(w + n * 64 + kk);
  u.h[1] = *(const v8h*)(w + n * 64 + kk + 16);
  return u.v;
}

// ---------------------------------------------------------------------------
// Pack x [B,T,50] fp32  ->  xh [T,B,64] f16 (zero-padded K)
// ---------------------------------------------------------------------------
__global__ __launch_bounds__(256) void cvt_x_kernel(const float* __restrict__ x,
                                                    _Float16* __restrict__ xh) {
  size_t i = (size_t)blockIdx.x * blockDim.x + threadIdx.x;
  const size_t total = (size_t)GRU_T * GRU_B * 64;
  if (i >= total) return;
  const int    c  = (int)(i & 63);
  const size_t bt = i >> 6;                 // = t*B + b
  const int    b  = (int)(bt % GRU_B);
  const int    t  = (int)(bt / GRU_B);
  float v = 0.0f;
  if (c < GRU_F) v = x[((size_t)b * GRU_T + t) * GRU_F + c];
  xh[i] = (_Float16)v;
}

// Pack weight [192,K] fp32 -> [192,64] f16 (zero-padded K)
__global__ __launch_bounds__(256) void cvt_w_kernel(const float* __restrict__ w,
                                                    _Float16* __restrict__ wh,
                                                    int K) {
  int i = blockIdx.x * blockDim.x + threadIdx.x;
  if (i >= 192 * 64) return;
  const int c = i & 63, r = i >> 6;
  wh[i] = (_Float16)((c < K) ? w[r * K + c] : 0.0f);
}

// ---------------------------------------------------------------------------
// Persistent fused GRU layer. One block = 16 batch rows, 4 waves.
// Wave w owns output columns [16w,16w+16) of each gate (r,z,n) so all gating
// is lane-local. n-gate keeps x-part / h-part accumulators separate because
// r multiplies only the h-part.
// ---------------------------------------------------------------------------
__global__ __launch_bounds__(128, 1) void gru_recur_kernel(
    const _Float16* __restrict__ xh,    // [T,B,64] f16 layer input (A matrix)
    const _Float16* __restrict__ wih,   // [192,64] f16
    const _Float16* __restrict__ whh,   // [192,64] f16
    const float*    __restrict__ bih,   // [192]
    const float*    __restrict__ bhh,   // [192]
    _Float16*       __restrict__ seq,   // [T,B,64] f16 output, or nullptr
    float*          __restrict__ hlast) // [B,64] fp32 final h, or nullptr
{
  __shared__ _Float16 hbuf[2][16 * 64];   // double-buffered h (f16, A layout)

  const int tid  = threadIdx.x;
  const int lane = tid & 31;
  const int wv   = tid >> 5;              // wave id 0..3
  const int b0   = blockIdx.x * 16;       // batch tile origin

  for (int i = tid; i < 16 * 64; i += 128) hbuf[0][i] = (_Float16)0.0f;

  const int nl = lane & 15;
  const int mh = (lane >> 4) << 3;        // row offset 0 or 8 (C/D layout)
  const int c  = (wv << 4) + nl;          // hidden column 0..63

  // Per-lane bias scalars (gate order r,z,n; columns c, 64+c, 128+c).
  const float br = bih[c] + bhh[c];       // r-gate biases fuse
  const float bz = bih[64 + c] + bhh[64 + c];
  const float bin = bih[128 + c], bhn = bhh[128 + c];

  // Weight B-fragments: VGPR-resident across the entire time loop.
  const v16h BiR0 = frag_b(wih,        (wv << 4),  0, lane);
  const v16h BiR1 = frag_b(wih,        (wv << 4), 32, lane);
  const v16h BiZ0 = frag_b(wih,  64 + (wv << 4),  0, lane);
  const v16h BiZ1 = frag_b(wih,  64 + (wv << 4), 32, lane);
  const v16h BiN0 = frag_b(wih, 128 + (wv << 4),  0, lane);
  const v16h BiN1 = frag_b(wih, 128 + (wv << 4), 32, lane);
  const v16h BhR0 = frag_b(whh,        (wv << 4),  0, lane);
  const v16h BhR1 = frag_b(whh,        (wv << 4), 32, lane);
  const v16h BhZ0 = frag_b(whh,  64 + (wv << 4),  0, lane);
  const v16h BhZ1 = frag_b(whh,  64 + (wv << 4), 32, lane);
  const v16h BhN0 = frag_b(whh, 128 + (wv << 4),  0, lane);
  const v16h BhN1 = frag_b(whh, 128 + (wv << 4), 32, lane);

  float hreg[8];                          // fp32 master copy of h (this lane)
#pragma unroll
  for (int i = 0; i < 8; ++i) hreg[i] = 0.0f;

  __syncthreads();

  for (int t = 0; t < GRU_T; ++t) {
    const int cur = t & 1;
    const _Float16* xbase = xh + ((size_t)t * GRU_B + b0) * 64;

    // Prefetch input tile for t+4 (global_prefetch_b8; 128 thr x 32B = 4KB).
    {
      const int tp = (t + 4 < GRU_T) ? t + 4 : t;
      const _Float16* pf = xh + ((size_t)tp * GRU_B + b0) * 64;
      __builtin_prefetch((const void*)(pf + tid * 16), 0, 0);
    }

    // A fragments: input row-block from global, h from LDS.
    v16h ax0 = frag_a(xbase, 0, lane);
    v16h ax1 = frag_a(xbase, 32, lane);
    v16h ah0 = frag_a((const _Float16*)hbuf[cur], 0, lane);
    v16h ah1 = frag_a((const _Float16*)hbuf[cur], 32, lane);

    v8f aR = {}; v8f aZ = {}; v8f aNX = {}; v8f aNH = {};
    // r gate: fused x-proj + h-proj
    aR  = wmma_f16(ax0, BiR0, aR);  aR  = wmma_f16(ax1, BiR1, aR);
    aR  = wmma_f16(ah0, BhR0, aR);  aR  = wmma_f16(ah1, BhR1, aR);
    // z gate
    aZ  = wmma_f16(ax0, BiZ0, aZ);  aZ  = wmma_f16(ax1, BiZ1, aZ);
    aZ  = wmma_f16(ah0, BhZ0, aZ);  aZ  = wmma_f16(ah1, BhZ1, aZ);
    // n gate: keep x-part and h-part separate (r scales only h-part)
    aNX = wmma_f16(ax0, BiN0, aNX); aNX = wmma_f16(ax1, BiN1, aNX);
    aNH = wmma_f16(ah0, BhN0, aNH); aNH = wmma_f16(ah1, BhN1, aNH);

    _Float16* hn = hbuf[cur ^ 1];
#pragma unroll
    for (int i = 0; i < 8; ++i) {
      const float rg = fast_sigmoid(aR[i] + br);
      const float zg = fast_sigmoid(aZ[i] + bz);
      const float ng = fast_tanh(aNX[i] + bin + rg * (aNH[i] + bhn));
      const float hv = ng + zg * (hreg[i] - ng);   // (1-z)*n + z*h
      hreg[i] = hv;
      const _Float16 hh = (_Float16)hv;
      hn[(mh + i) * 64 + c] = hh;
      if (seq) seq[((size_t)t * GRU_B + b0 + mh + i) * 64 + c] = hh;
    }
    __syncthreads();  // hbuf[cur^1] complete; hbuf[cur] reads done
  }

  if (hlast) {
#pragma unroll
    for (int i = 0; i < 8; ++i)
      hlast[(size_t)(b0 + mh + i) * 64 + c] = hreg[i];
  }
}

// ---------------------------------------------------------------------------
// Head: relu(h @ W1^T + b1) @ W2^T + b2   (trivial: 1 thread per batch row)
// ---------------------------------------------------------------------------
__global__ __launch_bounds__(256) void head_kernel(const float* __restrict__ hl,
                                                   const float* __restrict__ w1,
                                                   const float* __restrict__ b1,
                                                   const float* __restrict__ w2,
                                                   const float* __restrict__ b2,
                                                   float* __restrict__ out) {
  const int b = blockIdx.x * blockDim.x + threadIdx.x;
  if (b >= GRU_B) return;
  const float* h = hl + (size_t)b * 64;
  float acc = b2[0];
#pragma unroll 4
  for (int j = 0; j < 32; ++j) {
    float s = b1[j];
    const float* wr = w1 + j * 64;
#pragma unroll
    for (int k = 0; k < 64; ++k) s += h[k] * wr[k];
    acc += w2[j] * fmaxf(s, 0.0f);
  }
  out[b] = acc;
}

// ---------------------------------------------------------------------------
extern "C" void kernel_launch(void* const* d_in, const int* in_sizes, int n_in,
                              void* d_out, int out_size, void* d_ws, size_t ws_size,
                              hipStream_t stream) {
  (void)in_sizes; (void)n_in; (void)out_size; (void)ws_size;

  const float* x      = (const float*)d_in[0];
  const float* w_ih_0 = (const float*)d_in[1];
  const float* w_hh_0 = (const float*)d_in[2];
  const float* b_ih_0 = (const float*)d_in[3];
  const float* b_hh_0 = (const float*)d_in[4];
  const float* w_ih_1 = (const float*)d_in[5];
  const float* w_hh_1 = (const float*)d_in[6];
  const float* b_ih_1 = (const float*)d_in[7];
  const float* b_hh_1 = (const float*)d_in[8];
  const float* w_h1   = (const float*)d_in[9];
  const float* b_h1   = (const float*)d_in[10];
  const float* w_h2   = (const float*)d_in[11];
  const float* b_h2   = (const float*)d_in[12];
  float* out = (float*)d_out;

  char* ws = (char*)d_ws;
  size_t off = 0;
  auto take = [&](size_t bytes) {
    char* p = ws + off;
    off = (off + bytes + 255) & ~(size_t)255;
    return p;
  };
  _Float16* xh    = (_Float16*)take((size_t)GRU_T * GRU_B * 64 * sizeof(_Float16));
  _Float16* seq0  = (_Float16*)take((size_t)GRU_T * GRU_B * 64 * sizeof(_Float16));
  _Float16* wih0h = (_Float16*)take((size_t)192 * 64 * sizeof(_Float16));
  _Float16* whh0h = (_Float16*)take((size_t)192 * 64 * sizeof(_Float16));
  _Float16* wih1h = (_Float16*)take((size_t)192 * 64 * sizeof(_Float16));
  _Float16* whh1h = (_Float16*)take((size_t)192 * 64 * sizeof(_Float16));
  float*    hlast = (float*)take((size_t)GRU_B * 64 * sizeof(float));

  const size_t nx = (size_t)GRU_T * GRU_B * 64;
  cvt_x_kernel<<<(unsigned)((nx + 255) / 256), 256, 0, stream>>>(x, xh);
  const unsigned wgrid = (192 * 64 + 255) / 256;
  cvt_w_kernel<<<wgrid, 256, 0, stream>>>(w_ih_0, wih0h, GRU_F);
  cvt_w_kernel<<<wgrid, 256, 0, stream>>>(w_hh_0, whh0h, GRU_H);
  cvt_w_kernel<<<wgrid, 256, 0, stream>>>(w_ih_1, wih1h, GRU_H);
  cvt_w_kernel<<<wgrid, 256, 0, stream>>>(w_hh_1, whh1h, GRU_H);

  gru_recur_kernel<<<GRU_B / 16, 128, 0, stream>>>(xh, wih0h, whh0h, b_ih_0,
                                                   b_hh_0, seq0, nullptr);
  gru_recur_kernel<<<GRU_B / 16, 128, 0, stream>>>(seq0, wih1h, whh1h, b_ih_1,
                                                   b_hh_1, nullptr, hlast);

  head_kernel<<<1, 256, 0, stream>>>(hlast, w_h1, b_h1, w_h2, b_h2, out);
}
